// StackedElmanRNN_54915451846735
// MI455X (gfx1250) — matrix operand order repaired and checked by reading, not compiled, not run on hardware
//
#include <hip/hip_runtime.h>
#include <hip/hip_bf16.h>

// CDNA5 / gfx1250: wave32, WMMA f32_16x16x32_f16 (f16 in, f32 accumulate).
typedef __attribute__((ext_vector_type(16))) _Float16 v16h;
typedef __attribute__((ext_vector_type(8)))  float    v8f;

namespace {
constexpr int Bsz  = 2048;
constexpr int Tlen = 512;
constexpr int Edim = 64;
constexpr int H1   = 64;
constexpr int H2   = 32;
}

// Branch-free tanh: gfx1250 hardware v_tanh_f32 (TRANS unit).
__device__ __forceinline__ float fast_tanh(float x) {
#if __has_builtin(__builtin_amdgcn_tanhf)
  return __builtin_amdgcn_tanhf(x);
#elif __has_builtin(__builtin_amdgcn_tanh_f32)
  return __builtin_amdgcn_tanh_f32(x);
#else
  const float e = __builtin_amdgcn_exp2f(x * 2.885390081777927f);
  return 1.0f - 2.0f * __builtin_amdgcn_rcpf(e + 1.0f);
#endif
}

// LDS: weight fragments (hh1/ih2/hh2 read once into VGPRs; ih1 re-read each
// step on the off-critical-path xw pipeline) + per-wave C->A relayout scratch.
// One wave per block -> no cross-wave hazards, fences are s_wait_dscnt only.
struct __align__(32) Smem {
  _Float16 ih1[8 * 512];   // W_ih1^T fragments: stays in LDS (off-path use)
  _Float16 hh1[8 * 512];   // W_hh1^T: hoisted to VGPRs
  _Float16 ih2[4 * 512];   // W_ih2^T: hoisted to VGPRs
  _Float16 hh2[2 * 512];   // W_hh2^T: hoisted to VGPRs
  _Float16 h1s[2 * 512];   // h1_new relayout scratch (16x64)
  _Float16 h2s[512];       // h2_new relayout scratch (16x32)
  float    bias1[H1];      // b_ih1 + b_hh1
  float    bias2[H2];      // b_ih2 + b_hh2
  float    wcls[H2];       // classifier row (read only at capture)
};

// Stage W (N out-rows, K in-cols, row-major) into B-fragment layout:
// B[k][n] = W[n][k]; within a 32x16 fragment, lane = n + 16*bit3(k),
// half index = (k&7) + 8*bit4(k) (K packed 2 per VGPR).  Fragment (nt,kt) at (nt*KT+kt)*512.
__device__ __forceinline__ void stage_weight(const float* __restrict__ W,
                                             _Float16* dst, int N, int K,
                                             int tid, int nth) {
  const int KT = K >> 5;
  for (int i = tid; i < N * K; i += nth) {
    const int n = i / K, k = i - n * K;
    const int nt = n >> 4, kt = k >> 5;
    const int nf = n & 15, kf = k & 31;
    const int lane = nf + (((kf >> 3) & 1) << 4);
    dst[((nt * KT + kt) << 9) + lane * 16 + ((kf & 7) + ((kf >> 4) << 3))] = (_Float16)W[i];
  }
}

__device__ __forceinline__ v16h load_bfrag(const _Float16* frag, int lane) {
  return *(const v16h*)(frag + lane * 16);
}

// Build one 16x32 f16 A-fragment from a contiguous f32 row slice.
// ISA A-layout: this lane's halves 0..7 <- src[0..7], 8..15 <- src[16..23]
// (caller offsets src by +8 floats for the upper lane group).
__device__ __forceinline__ v16h gather_frag(const float* __restrict__ src) {
  const float4 a = ((const float4*)src)[0];
  const float4 b = ((const float4*)src)[1];
  const float4 c = ((const float4*)(src + 16))[0];
  const float4 d = ((const float4*)(src + 16))[1];
  v16h r;
  r[0]  = (_Float16)a.x; r[1]  = (_Float16)a.y;
  r[2]  = (_Float16)a.z; r[3]  = (_Float16)a.w;
  r[4]  = (_Float16)b.x; r[5]  = (_Float16)b.y;
  r[6]  = (_Float16)b.z; r[7]  = (_Float16)b.w;
  r[8]  = (_Float16)c.x; r[9]  = (_Float16)c.y;
  r[10] = (_Float16)c.z; r[11] = (_Float16)c.w;
  r[12] = (_Float16)d.x; r[13] = (_Float16)d.y;
  r[14] = (_Float16)d.z; r[15] = (_Float16)d.w;
  return r;
}

__device__ __forceinline__ v8f wmma_acc(v16h a, v16h b, v8f c) {
  return __builtin_amdgcn_wmma_f32_16x16x32_f16(
      /*neg_a=*/false, a, /*neg_b=*/false, b,
      /*c_mod=*/(short)0, c, /*reuse_a=*/false, /*reuse_b=*/false);
}

__device__ __forceinline__ v8f splat8(float v) {
  v8f r;
#pragma unroll
  for (int i = 0; i < 8; ++i) r[i] = v;
  return r;
}

// Scatter a C/D 16x16 f32 result tile (VGPR r: M=r / r+8; N = lane&15) into
// the A-fragment layout in LDS scratch, converting to f16.  Target mapping:
// element (m,k): lane' = m + 16*bit3(k), half = (k&7) + 8*bit4(k), frag = k>>5.
__device__ __forceinline__ void store_cfrag(_Float16* scratch, int ntbase,
                                            int lane, const float (&res)[8]) {
  const int n    = lane & 15;
  const int k    = ntbase + n;
  const int f    = k >> 5;
  const int kf   = k & 31;
  const int lgrp = (kf >> 3) & 1;
  const int h    = (kf & 7) + ((kf >> 4) << 3);
  const int mb   = (lane >> 4) << 3;   // C layout: upper lanes hold M+8
  _Float16* p = scratch + (f << 9) + h;
#pragma unroll
  for (int r = 0; r < 8; ++r)
    p[(mb + r + (lgrp << 4)) * 16] = (_Float16)res[r];
}

__global__ void __launch_bounds__(32, 1)
elman_rnn_kernel(const int* __restrict__ tokens, const int* __restrict__ lengths,
                 const float* __restrict__ emb,
                 const float* __restrict__ W_ih1, const float* __restrict__ W_hh1,
                 const float* __restrict__ b_ih1, const float* __restrict__ b_hh1,
                 const float* __restrict__ W_ih2, const float* __restrict__ W_hh2,
                 const float* __restrict__ b_ih2, const float* __restrict__ b_hh2,
                 const float* __restrict__ W_cls, const float* __restrict__ b_cls,
                 float* __restrict__ out_logits, float* __restrict__ out_final) {
  __shared__ Smem smem;
  const int lane = threadIdx.x;         // one wave32 per block
  const int tid = lane, nth = 32;

  stage_weight(W_ih1, smem.ih1, H1, Edim, tid, nth);
  stage_weight(W_hh1, smem.hh1, H1, H1,   tid, nth);
  stage_weight(W_ih2, smem.ih2, H2, H1,   tid, nth);
  stage_weight(W_hh2, smem.hh2, H2, H2,   tid, nth);
  for (int i = tid; i < H1; i += nth) smem.bias1[i] = b_ih1[i] + b_hh1[i];
  for (int i = tid; i < H2; i += nth) smem.bias2[i] = b_ih2[i] + b_hh2[i];
  for (int i = tid; i < H2; i += nth) smem.wcls[i]  = W_cls[i];
  __syncthreads();

  // Only the LATENCY-CRITICAL operands live in VGPRs (14 frags = 112 VGPRs):
  // the recurrent-path weights.  W_ih1 stays in LDS -- its consumer (the
  // pipelined x@W_ih1 precompute) is off the critical path.
  v16h Bhh1[8], Bih2[4], Bhh2[2];
#pragma unroll
  for (int i = 0; i < 8; ++i) Bhh1[i] = load_bfrag(smem.hh1 + i * 512, lane);
#pragma unroll
  for (int i = 0; i < 4; ++i) Bih2[i] = load_bfrag(smem.ih2 + i * 512, lane);
#pragma unroll
  for (int i = 0; i < 2; ++i) Bhh2[i] = load_bfrag(smem.hh2 + i * 512, lane);

  const int m   = lane & 15;            // batch row within tile
  const int hi  = lane >> 4;            // lane group (K-split of A layout)
  const int row = blockIdx.x * 16 + m;
  const int* __restrict__ tokrow = tokens + (size_t)row * Tlen;
  const int lastT = lengths[row] - 1;   // capture step (post-update)
  const float bcls = b_cls[0];

  // Persistent bias splats as WMMA C-operand seeds.  WMMA is three-address
  // (D != C), so these registers are never clobbered -> zero per-step movs.
  v8f bias1c[4], bias2c[2];
#pragma unroll
  for (int nt = 0; nt < 4; ++nt) bias1c[nt] = splat8(smem.bias1[nt * 16 + m]);
#pragma unroll
  for (int nt = 0; nt < 2; ++nt) bias2c[nt] = splat8(smem.bias2[nt * 16 + m]);

  v16h h1a0 = {}, h1a1 = {}, h2a = {};  // recurrent state, A-fragment layout

  // Cross-step software pipeline: xw[nt] = bias1 + x_t @ W_ih1^T for the
  // step about to run.  Computed one step ahead, off the critical path.
  v8f xw[4];
  {
    const int tk0 = tokrow[0];
    const float* e0 = emb + (size_t)tk0 * Edim + hi * 8;
    const v16h x0 = gather_frag(e0);
    const v16h x1 = gather_frag(e0 + 32);
#pragma unroll
    for (int nt = 0; nt < 4; ++nt) {
      v8f a = wmma_acc(x0, load_bfrag(smem.ih1 + (nt * 2 + 0) * 512, lane),
                       bias1c[nt]);
      xw[nt] = wmma_acc(x1, load_bfrag(smem.ih1 + (nt * 2 + 1) * 512, lane), a);
    }
  }

  for (int t = 0; t < Tlen; ++t) {
    // ---- Layer-2 partial: b2 + h2@W_hh2^T depends only on h2a (ready since
    // last step) -> issue it now, concurrent with layer 1.  This shortens the
    // post-h1' chain of layer 2 from 3 WMMAs to 2.
    v8f h2p[2];
#pragma unroll
    for (int nt = 0; nt < 2; ++nt)
      h2p[nt] = wmma_acc(h2a, Bhh2[nt], bias2c[nt]);

    // ---- Layer 1 (critical chain = 2 WMMAs): tanh(xw + h1@W_hh1^T).
#pragma unroll
    for (int nt = 0; nt < 4; ++nt) {
      v8f acc = wmma_acc(h1a0, Bhh1[nt * 2 + 0], xw[nt]); // bias+x already in xw
      acc     = wmma_acc(h1a1, Bhh1[nt * 2 + 1], acc);
      float res[8];
#pragma unroll
      for (int r = 0; r < 8; ++r) res[r] = fast_tanh(acc[r]);
      store_cfrag(smem.h1s, nt * 16, lane, res);
    }

    // ---- Off-path: gather x(t+1) and precompute next step's xw while the
    // layer-1 tanh/relayout drains.  Uniform branch -> EXEC stays all-ones.
    v8f xwn[4];
    if (t + 1 < Tlen) {
      const int tk2 = tokrow[t + 1];
      const float* e2 = emb + (size_t)tk2 * Edim + hi * 8;
      const v16h xn0 = gather_frag(e2);
      const v16h xn1 = gather_frag(e2 + 32);
#pragma unroll
      for (int nt = 0; nt < 4; ++nt) {
        v8f a = wmma_acc(xn0, load_bfrag(smem.ih1 + (nt*2+0)*512, lane),
                         bias1c[nt]);
        xwn[nt] = wmma_acc(xn1, load_bfrag(smem.ih1 + (nt*2+1)*512, lane), a);
      }
    } else {
#pragma unroll
      for (int nt = 0; nt < 4; ++nt) xwn[nt] = bias1c[nt];  // rare tail copy
    }

    asm volatile("s_wait_dscnt 0x0" ::: "memory");  // single-wave relayout fence
    h1a0 = *(const v16h*)(smem.h1s + lane * 16);        // new h1 state (A layout)
    h1a1 = *(const v16h*)(smem.h1s + 512 + lane * 16);

    // ---- Layer 2 completion (chain = 2 WMMAs): tanh(h2p + h1'@W_ih2^T).
#pragma unroll
    for (int nt = 0; nt < 2; ++nt) {
      v8f acc = wmma_acc(h1a0, Bih2[nt * 2 + 0], h2p[nt]);
      acc     = wmma_acc(h1a1, Bih2[nt * 2 + 1], acc);
      float res[8];
#pragma unroll
      for (int r = 0; r < 8; ++r) res[r] = fast_tanh(acc[r]);
      store_cfrag(smem.h2s, nt * 16, lane, res);
    }
    asm volatile("s_wait_dscnt 0x0" ::: "memory");
    h2a = *(const v16h*)(smem.h2s + lane * 16);          // new h2 state

    // ---- Capture final hidden + logit at t == lengths[row]-1 (post-update).
    if (t == lastT) {
      float* fh = out_final + (size_t)row * H2;
      float acc = 0.f;
#pragma unroll
      for (int h = 0; h < 16; ++h) {
        const int k = hi * 8 + (h < 8 ? h : h + 8);
        const float v = (float)h2a[h];
        fh[k] = v;
        acc += v * smem.wcls[k];
      }
      acc += __shfl_xor(acc, 16, 32);   // partner lane (same row) is active too
      if (hi == 0) out_logits[row] = acc + bcls;
    }

#pragma unroll
    for (int nt = 0; nt < 4; ++nt) xw[nt] = xwn[nt];
  }
}

extern "C" void kernel_launch(void* const* d_in, const int* in_sizes, int n_in,
                              void* d_out, int out_size, void* d_ws,
                              size_t ws_size, hipStream_t stream) {
  (void)in_sizes; (void)n_in; (void)out_size; (void)d_ws; (void)ws_size;
  const int*   tokens  = (const int*)d_in[0];
  const int*   lengths = (const int*)d_in[1];
  const float* emb     = (const float*)d_in[2];
  const float* W_ih1   = (const float*)d_in[3];
  const float* W_hh1   = (const float*)d_in[4];
  const float* b_ih1   = (const float*)d_in[5];
  const float* b_hh1   = (const float*)d_in[6];
  const float* W_ih2   = (const float*)d_in[7];
  const float* W_hh2   = (const float*)d_in[8];
  const float* b_ih2   = (const float*)d_in[9];
  const float* b_hh2   = (const float*)d_in[10];
  const float* W_cls   = (const float*)d_in[11];
  const float* b_cls   = (const float*)d_in[12];

  float* out        = (float*)d_out;
  float* out_logits = out;              // (B,)
  float* out_final  = out + Bsz;        // (B, H2) flat

  dim3 grid(Bsz / 16), block(32);
  hipLaunchKernelGGL(elman_rnn_kernel, grid, block, 0, stream,
                     tokens, lengths, emb, W_ih1, W_hh1, b_ih1, b_hh1,
                     W_ih2, W_hh2, b_ih2, b_hh2, W_cls, b_cls,
                     out_logits, out_final);
}
